// LinearMoEBlock_51883204935740
// MI455X (gfx1250) — matrix operand order repaired and checked by reading, not compiled, not run on hardware
//
#include <hip/hip_runtime.h>
#include <hip/hip_bf16.h>

// ---------------------------------------------------------------------------
// MI455X (gfx1250) implementation. All heavy GEMMs run on
// v_wmma_f32_16x16x32_bf16 (wave32 WMMA), operands staged through LDS with
// double buffering; fragment gathers are 2x ds_load_b128 per v16bf matching
// the CDNA5 16-bit A/B VGPR layouts (CDNA5 ISA §7.12.2).
// Global->LDS copies use GLOBAL_LOAD_ASYNC_TO_LDS_B128 (ASYNCcnt) when the
// toolchain exposes the builtin; otherwise fall back to b128 load/store.
// ---------------------------------------------------------------------------

typedef __attribute__((ext_vector_type(16))) __bf16 v16bf;
typedef __attribute__((ext_vector_type(8)))  float  v8f;

#if __has_builtin(__builtin_amdgcn_global_load_async_to_lds_b128) && \
    __has_builtin(__builtin_amdgcn_s_wait_asynccnt)
#define USE_ASYNC_COPY 1
// Builtin signature (from hipcc diagnostics): (int4 AS1*, int4 AS3*, Ii, Ii)
typedef int v4i_ __attribute__((vector_size(16)));
typedef __attribute__((address_space(1))) v4i_ as1_v4i;
typedef __attribute__((address_space(3))) v4i_ as3_v4i;
#else
#define USE_ASYNC_COPY 0
#endif

#define DEV static __device__ __forceinline__

DEV unsigned short f2bf(float x) {
  union { float f; unsigned int u; } v; v.f = x;
  unsigned int u = v.u;
  unsigned int r = (u + 0x7FFFu + ((u >> 16) & 1u)) >> 16;  // RNE truncate
  return (unsigned short)r;
}
DEV float bf2f(unsigned short b) {
  union { unsigned int u; float f; } v; v.u = ((unsigned int)b) << 16; return v.f;
}

DEV v16bf frag16(const unsigned short* p0, const unsigned short* p1) {
  union { v16bf v; uint4 q[2]; } u;
  u.q[0] = *(const uint4*)p0;   // -> ds_load_b128
  u.q[1] = *(const uint4*)p1;   // -> ds_load_b128
  return u.v;
}

DEV void blockReduce2(float& s, float& s2) {
  #pragma unroll
  for (int o = 16; o; o >>= 1) { s += __shfl_xor(s, o, 32); s2 += __shfl_xor(s2, o, 32); }
  __shared__ float sh[16];
  const int lane = threadIdx.x & 31, wv = threadIdx.x >> 5;
  if (lane == 0) { sh[wv] = s; sh[8 + wv] = s2; }
  __syncthreads();
  float a = 0.f, b = 0.f;
  #pragma unroll
  for (int i = 0; i < 8; ++i) { a += sh[i]; b += sh[8 + i]; }
  s = a; s2 = b;
}

// ---------------------------------------------------------------------------
// Generic TN WMMA GEMM:  C[M][N] = epilogue( sum_k A[m][k] * Bt[n][k] )
// A row-major bf16 (lda), Bt = B^T row-major bf16 (ldb). Block tile 128x128,
// K-step 32, 8 waves of 64x32 each (4x2 WMMA tiles). Batched via blockIdx.z
// with split strides: z1 = z/ZD, z2 = z%ZD.
// act: 0 none, 1 elu(x)+1, 2 exact GELU (bias applied BEFORE activation).
// ---------------------------------------------------------------------------
__global__ __launch_bounds__(256)
void wmma_gemm_bf16(const unsigned short* __restrict__ A, int lda, long long aS1, long long aS2,
                    const unsigned short* __restrict__ Bt, int ldb, long long bS1, long long bS2,
                    void* __restrict__ Cv, int ldc, long long cS1, long long cS2,
                    int M, int N, int K, int ZD,
                    const float* __restrict__ bias,
                    const float* __restrict__ resid,
                    const float* __restrict__ rowscale, long long rsS1, long long rsS2,
                    const float* __restrict__ gate, int gstride,
                    int act, int bf16out, int transout)
{
  __shared__ __align__(16) unsigned short As[2][128 * 40];  // 32 K + 8 pad
  __shared__ __align__(16) unsigned short Bs[2][128 * 40];

  const int tid  = threadIdx.x;
  const int lane = tid & 31;
  const int wave = tid >> 5;
  const int lrow = lane & 15;
  const int hi   = lane >> 4;                 // 0: K-low half, 1: K-high half
  const int wm   = (wave >> 2) * 64;
  const int wn   = (wave & 3) * 32;
  const int m0   = blockIdx.y * 128;
  const int n0   = blockIdx.x * 128;
  const int z    = blockIdx.z;
  const int z1   = z / ZD, z2 = z % ZD;

  A  += (long long)z1 * aS1 + (long long)z2 * aS2;
  Bt += (long long)z1 * bS1 + (long long)z2 * bS2;
  const float* rs = rowscale ? rowscale + (long long)z1 * rsS1 + (long long)z2 * rsS2 : nullptr;

  v8f acc[4][2];
  #pragma unroll
  for (int i = 0; i < 4; ++i)
    #pragma unroll
    for (int j = 0; j < 2; ++j)
      #pragma unroll
      for (int e = 0; e < 8; ++e) acc[i][j][e] = 0.f;

  // Issue the global->LDS tile copy for one K-step (async when available).
  auto issueTiles = [&](int kk, int buf) {
    #pragma unroll
    for (int i = 0; i < 2; ++i) {
      const int c  = tid + i * 256;           // 512 16B chunks per tile
      const int r  = c >> 2;
      const int cc = (c & 3) << 3;
      const unsigned short* ga = A  + (long long)(m0 + r) * lda + kk + cc;
      const unsigned short* gb = Bt + (long long)(n0 + r) * ldb + kk + cc;
      unsigned short* la = &As[buf][r * 40 + cc];
      unsigned short* lb = &Bs[buf][r * 40 + cc];
#if USE_ASYNC_COPY
      __builtin_amdgcn_global_load_async_to_lds_b128(
          (as1_v4i*)(unsigned short*)ga, (as3_v4i*)la, 0, 0);
      __builtin_amdgcn_global_load_async_to_lds_b128(
          (as1_v4i*)(unsigned short*)gb, (as3_v4i*)lb, 0, 0);
#else
      *(uint4*)la = *(const uint4*)ga;        // global_load_b128 -> ds_store_b128
      *(uint4*)lb = *(const uint4*)gb;
#endif
    }
  };

  const int nsteps = K >> 5;
  issueTiles(0, 0);
#if USE_ASYNC_COPY
  __builtin_amdgcn_s_wait_asynccnt(0);
#endif
  __syncthreads();

  for (int s = 0; s < nsteps; ++s) {
    const int buf = s & 1;
    if (s + 1 < nsteps) issueTiles((s + 1) << 5, buf ^ 1);
    if (s + 2 < nsteps) {                       // global_prefetch_b8 two steps ahead
      __builtin_prefetch((const void*)(A  + (long long)(m0 + (tid >> 1)) * lda + ((s + 2) << 5) + ((tid & 1) << 4)), 0, 0);
      __builtin_prefetch((const void*)(Bt + (long long)(n0 + (tid >> 1)) * ldb + ((s + 2) << 5) + ((tid & 1) << 4)), 0, 0);
    }

    // A frag: lanes 0-15 K{0..7,16..23}, lanes 16-31 K{8..15,24..31} of row M=lane%16
    v16bf af[4], bfr[2];
    #pragma unroll
    for (int mt = 0; mt < 4; ++mt) {
      const unsigned short* p = &As[buf][(wm + mt * 16 + lrow) * 40 + hi * 8];
      af[mt] = frag16(p, p + 16);
    }
    // B frag: lanes 0-15 K{0..7,16..23}, lanes 16-31 K{8..15,24..31} of col N=lane%16
    #pragma unroll
    for (int nt = 0; nt < 2; ++nt) {
      const unsigned short* p = &Bs[buf][(wn + nt * 16 + lrow) * 40 + hi * 8];
      bfr[nt] = frag16(p, p + 16);
    }
    #pragma unroll
    for (int mt = 0; mt < 4; ++mt)
      #pragma unroll
      for (int nt = 0; nt < 2; ++nt)
        acc[mt][nt] = __builtin_amdgcn_wmma_f32_16x16x32_bf16(
            false, af[mt], false, bfr[nt], (short)0, acc[mt][nt], false, false);

#if USE_ASYNC_COPY
    __builtin_amdgcn_s_wait_asynccnt(0);        // drain this wave's async tile writes
#endif
    __syncthreads();                            // make LDS tile visible block-wide
  }

  unsigned short* o16 = nullptr; float* o32 = nullptr;
  if (bf16out) o16 = (unsigned short*)Cv + (long long)z1 * cS1 + (long long)z2 * cS2;
  else         o32 = (float*)Cv          + (long long)z1 * cS1 + (long long)z2 * cS2;

  #pragma unroll
  for (int mt = 0; mt < 4; ++mt) {
    #pragma unroll
    for (int nt = 0; nt < 2; ++nt) {
      #pragma unroll
      for (int i = 0; i < 8; ++i) {
        const int mm = m0 + wm + mt * 16 + hi * 8 + i;   // VGPR i: M=i (lo) / 8+i (hi)
        const int nn = n0 + wn + nt * 16 + lrow;
        float vv = acc[mt][nt][i];
        if (bias) vv += bias[nn];
        if (act == 1)      vv = vv > 0.f ? vv + 1.f : __expf(vv);          // elu(x)+1
        else if (act == 2) vv = 0.5f * vv * (1.f + erff(vv * 0.70710678118654752f));
        if (rs) vv *= rs[mm];
        if (resid) vv += resid[(long long)mm * ldc + nn];
        if (gate) {
          o32[(long long)mm * ldc + nn] += gate[(long long)mm * gstride] * vv;
        } else if (bf16out) {
          if (transout) o16[(long long)nn * ldc + mm] = f2bf(vv);
          else          o16[(long long)mm * ldc + nn] = f2bf(vv);
        } else {
          o32[(long long)mm * ldc + nn] = vv;
        }
      }
    }
  }
}

// ---------------------------------------------------------------------------
// fp32 [Kin][Nout] -> bf16 [Nout][Kin] (TN layout for the GEMM), batched on z
// ---------------------------------------------------------------------------
__global__ __launch_bounds__(256)
void transpose_bf16(const float* __restrict__ W, unsigned short* __restrict__ WT,
                    int Kin, int Nout)
{
  const long long total = (long long)Kin * Nout;
  const long long idx = (long long)blockIdx.x * 256 + threadIdx.x;
  W  += (long long)blockIdx.z * total;
  WT += (long long)blockIdx.z * total;
  if (idx < total) {
    const int kk = (int)(idx / Nout), nn = (int)(idx % Nout);
    WT[(long long)nn * Kin + kk] = f2bf(W[idx]);
  }
}

// LayerNorm over D=1024, one block per row, fp32 in -> bf16 out
__global__ __launch_bounds__(256)
void ln_bf16(const float* __restrict__ x, const float* __restrict__ g,
             const float* __restrict__ b, unsigned short* __restrict__ y)
{
  const int row = blockIdx.x;
  const float4 v4 = ((const float4*)(x + (long long)row * 1024))[threadIdx.x];
  float s  = v4.x + v4.y + v4.z + v4.w;
  float s2 = v4.x * v4.x + v4.y * v4.y + v4.z * v4.z + v4.w * v4.w;
  blockReduce2(s, s2);
  const float mean = s * (1.f / 1024.f);
  const float var  = s2 * (1.f / 1024.f) - mean * mean;
  const float rstd = rsqrtf(var + 1e-5f);
  const int c = threadIdx.x * 4;
  ushort4 o;
  o.x = f2bf((v4.x - mean) * rstd * g[c + 0] + b[c + 0]);
  o.y = f2bf((v4.y - mean) * rstd * g[c + 1] + b[c + 1]);
  o.z = f2bf((v4.z - mean) * rstd * g[c + 2] + b[c + 2]);
  o.w = f2bf((v4.w - mean) * rstd * g[c + 3] + b[c + 3]);
  *(ushort4*)(y + (long long)row * 1024 + c) = o;
}

// ksum[b*1024 + h*128+d] = sum_t khT[h*128+d][b*2048+t] + 1e-6 ; 2048 waves
__global__ __launch_bounds__(256)
void ksum_kernel(const unsigned short* __restrict__ khT, float* __restrict__ ksum)
{
  const int w = (blockIdx.x * 256 + threadIdx.x) >> 5;
  const int lane = threadIdx.x & 31;
  const int b = w >> 10, hd = w & 1023;
  const unsigned short* p = khT + (long long)hd * 4096 + b * 2048;
  float s = 0.f;
  for (int i = lane; i < 2048; i += 32) s += bf2f(p[i]);
  #pragma unroll
  for (int o = 16; o; o >>= 1) s += __shfl_xor(s, o, 32);
  if (lane == 0) ksum[b * 1024 + hd] = s + 1e-6f;
}

// rden[(b*8+h)*2048 + t] = 1 / (qh[b*2048+t][h*128 : +128] . ksum[b,h,:]) ; 32768 waves
__global__ __launch_bounds__(256)
void denom_kernel(const unsigned short* __restrict__ qh, const float* __restrict__ ksum,
                  float* __restrict__ rden)
{
  const int w = (blockIdx.x * 256 + threadIdx.x) >> 5;
  const int lane = threadIdx.x & 31;
  const int b = w >> 14, h = (w >> 11) & 7, t = w & 2047;
  const unsigned short* q = qh + (long long)(b * 2048 + t) * 1024 + h * 128;
  const float* ks = ksum + (b * 8 + h) * 128;
  float s = 0.f;
  #pragma unroll
  for (int d = lane; d < 128; d += 32) s += bf2f(q[d]) * ks[d];
  #pragma unroll
  for (int o = 16; o; o >>= 1) s += __shfl_xor(s, o, 32);
  if (lane == 0) rden[(b * 8 + h) * 2048 + t] = 1.f / s;
}

// Noisy top-2 router: one wave per token. gates[n][e] = softmax_e if e in top2 else 0
__global__ __launch_bounds__(256)
void router_kernel(const unsigned short* __restrict__ xm, const float* __restrict__ Wr,
                   const float* __restrict__ br, const float* __restrict__ noise,
                   float* __restrict__ gates)
{
  const int w = (blockIdx.x * 256 + threadIdx.x) >> 5;   // token id
  const int lane = threadIdx.x & 31;
  const unsigned short* xr = xm + (long long)w * 1024;
  float a[8] = {0, 0, 0, 0, 0, 0, 0, 0};
  for (int d = lane; d < 1024; d += 32) {
    const float xv = bf2f(xr[d]);
    const float* wr = Wr + d * 8;
    #pragma unroll
    for (int e = 0; e < 8; ++e) a[e] += xv * wr[e];
  }
  #pragma unroll
  for (int e = 0; e < 8; ++e)
    #pragma unroll
    for (int o = 16; o; o >>= 1) a[e] += __shfl_xor(a[e], o, 32);
  if (lane == 0) {
    float lg[8], mx = -1e30f;
    #pragma unroll
    for (int e = 0; e < 8; ++e) { lg[e] = a[e] + br[e] + noise[w * 8 + e]; mx = fmaxf(mx, lg[e]); }
    float sum = 0.f;
    #pragma unroll
    for (int e = 0; e < 8; ++e) { lg[e] = expf(lg[e] - mx); sum += lg[e]; }
    const float inv = 1.f / sum;
    #pragma unroll
    for (int e = 0; e < 8; ++e) lg[e] *= inv;
    int i1 = 0;
    #pragma unroll
    for (int e = 1; e < 8; ++e) if (lg[e] > lg[i1]) i1 = e;
    int i2 = -1;
    #pragma unroll
    for (int e = 0; e < 8; ++e) if (e != i1 && (i2 < 0 || lg[e] > lg[i2])) i2 = e;
    #pragma unroll
    for (int e = 0; e < 8; ++e) gates[w * 8 + e] = (e == i1 || e == i2) ? lg[e] : 0.f;
  }
}

__global__ __launch_bounds__(256)
void zero_f32(float* __restrict__ p, int n)
{
  const int i = blockIdx.x * 256 + threadIdx.x;
  if (i < n) p[i] = 0.f;
}

// out = q2 + LN(mo + xm) * gn + bn
__global__ __launch_bounds__(256)
void final_kernel(const float* __restrict__ mo, const unsigned short* __restrict__ xm,
                  const float* __restrict__ q2, const float* __restrict__ gn,
                  const float* __restrict__ bn, float* __restrict__ out)
{
  const int row = blockIdx.x;
  const int c = threadIdx.x * 4;
  const float4 m4 = ((const float4*)(mo + (long long)row * 1024))[threadIdx.x];
  const unsigned short* xr = xm + (long long)row * 1024 + c;
  const float t0 = m4.x + bf2f(xr[0]);
  const float t1 = m4.y + bf2f(xr[1]);
  const float t2 = m4.z + bf2f(xr[2]);
  const float t3 = m4.w + bf2f(xr[3]);
  float s  = t0 + t1 + t2 + t3;
  float s2 = t0 * t0 + t1 * t1 + t2 * t2 + t3 * t3;
  blockReduce2(s, s2);
  const float mean = s * (1.f / 1024.f);
  const float var  = s2 * (1.f / 1024.f) - mean * mean;
  const float rstd = rsqrtf(var + 1e-5f);
  const float4 q4 = ((const float4*)(q2 + (long long)row * 1024))[threadIdx.x];
  float4 o;
  o.x = q4.x + (t0 - mean) * rstd * gn[c + 0] + bn[c + 0];
  o.y = q4.y + (t1 - mean) * rstd * gn[c + 1] + bn[c + 1];
  o.z = q4.z + (t2 - mean) * rstd * gn[c + 2] + bn[c + 2];
  o.w = q4.w + (t3 - mean) * rstd * gn[c + 3] + bn[c + 3];
  ((float4*)(out + (long long)row * 1024))[threadIdx.x] = o;
}

// ---------------------------------------------------------------------------
extern "C" void kernel_launch(void* const* d_in, const int* in_sizes, int n_in,
                              void* d_out, int out_size, void* d_ws, size_t ws_size,
                              hipStream_t stream)
{
  (void)in_sizes; (void)n_in; (void)out_size; (void)ws_size;
  const float* V  = (const float*)d_in[0];
  const float* Kx = (const float*)d_in[1];
  const float* Q  = (const float*)d_in[2];
  const float* gv = (const float*)d_in[3];
  const float* bv = (const float*)d_in[4];
  const float* gk = (const float*)d_in[5];
  const float* bk = (const float*)d_in[6];
  const float* gq = (const float*)d_in[7];
  const float* bq = (const float*)d_in[8];
  const float* Wq = (const float*)d_in[9];
  const float* Wk = (const float*)d_in[10];
  const float* Wv = (const float*)d_in[11];
  const float* Wo = (const float*)d_in[12];
  const float* bo = (const float*)d_in[13];
  const float* gm = (const float*)d_in[14];
  const float* bm = (const float*)d_in[15];
  const float* Wr = (const float*)d_in[16];
  const float* br = (const float*)d_in[17];
  const float* noise = (const float*)d_in[18];
  const float* W1 = (const float*)d_in[19];
  const float* b1 = (const float*)d_in[20];
  const float* W2 = (const float*)d_in[21];
  const float* b2 = (const float*)d_in[22];
  const float* gn = (const float*)d_in[23];
  const float* bn = (const float*)d_in[24];
  float* out = (float*)d_out;

  // Workspace layout (with deliberate overlays; total ~137.5 MB).
  char* ws = (char*)d_ws;
  const size_t MB = 1024ull * 1024ull;
  unsigned short* WqT  = (unsigned short*)(ws + 0 * MB);    // 2 MB
  unsigned short* WkT  = (unsigned short*)(ws + 2 * MB);    // 2 MB
  unsigned short* WvT  = (unsigned short*)(ws + 4 * MB);    // 2 MB
  unsigned short* WoT  = (unsigned short*)(ws + 6 * MB);    // 2 MB
  unsigned short* W1T  = (unsigned short*)(ws + 8 * MB);    // 32 MB
  unsigned short* W2T  = (unsigned short*)(ws + 40 * MB);   // 32 MB
  unsigned short* qn   = (unsigned short*)(ws + 72 * MB);   // 8 MB
  unsigned short* kn   = (unsigned short*)(ws + 80 * MB);   // 8 MB
  unsigned short* hbuf = (unsigned short*)(ws + 72 * MB);   // 16 MB, overlays qn+kn (dead)
  unsigned short* vn   = (unsigned short*)(ws + 88 * MB);   // 8 MB
  unsigned short* attn = (unsigned short*)(ws + 88 * MB);   // overlays vn (dead)
  unsigned short* qh   = (unsigned short*)(ws + 96 * MB);   // 8 MB
  unsigned short* khT  = (unsigned short*)(ws + 104 * MB);  // 8 MB
  float*          mo   = (float*)(ws + 96 * MB);            // 16 MB, overlays qh+khT (dead)
  unsigned short* vhT  = (unsigned short*)(ws + 112 * MB);  // 8 MB
  unsigned short* xm   = (unsigned short*)(ws + 112 * MB);  // overlays vhT (dead)
  float*          q2   = (float*)(ws + 120 * MB);           // 16 MB
  unsigned short* kvT  = (unsigned short*)(ws + 136 * MB);  // 512 KB
  float*          ksum = (float*)(ws + 137 * MB);           // 8 KB
  float*          rden = (float*)(ws + 137 * MB + 64 * 1024);   // 128 KB
  float*          gts  = (float*)(ws + 137 * MB + 256 * 1024);  // 128 KB

  // 1) weights -> bf16 TN layout
  transpose_bf16<<<dim3(4096, 1, 1), 256, 0, stream>>>(Wq, WqT, 1024, 1024);
  transpose_bf16<<<dim3(4096, 1, 1), 256, 0, stream>>>(Wk, WkT, 1024, 1024);
  transpose_bf16<<<dim3(4096, 1, 1), 256, 0, stream>>>(Wv, WvT, 1024, 1024);
  transpose_bf16<<<dim3(4096, 1, 1), 256, 0, stream>>>(Wo, WoT, 1024, 1024);
  transpose_bf16<<<dim3(8192, 1, 8), 256, 0, stream>>>(W1, W1T, 1024, 2048);
  transpose_bf16<<<dim3(8192, 1, 8), 256, 0, stream>>>(W2, W2T, 2048, 1024);

  // 2) pre-attention LayerNorms -> bf16
  ln_bf16<<<4096, 256, 0, stream>>>(Q,  gq, bq, qn);
  ln_bf16<<<4096, 256, 0, stream>>>(Kx, gk, bk, kn);
  ln_bf16<<<4096, 256, 0, stream>>>(V,  gv, bv, vn);

  // 3) projections (elu+1 on q,k). k/v stored transposed [H*DH][B*T] for TN reuse.
  wmma_gemm_bf16<<<dim3(8, 32, 1), 256, 0, stream>>>(qn, 1024, 0, 0, WqT, 1024, 0, 0,
      qh, 1024, 0, 0, 4096, 1024, 1024, 1, nullptr, nullptr, nullptr, 0, 0, nullptr, 0, 1, 1, 0);
  wmma_gemm_bf16<<<dim3(8, 32, 1), 256, 0, stream>>>(kn, 1024, 0, 0, WkT, 1024, 0, 0,
      khT, 4096, 0, 0, 4096, 1024, 1024, 1, nullptr, nullptr, nullptr, 0, 0, nullptr, 0, 1, 1, 1);
  wmma_gemm_bf16<<<dim3(8, 32, 1), 256, 0, stream>>>(vn, 1024, 0, 0, WvT, 1024, 0, 0,
      vhT, 4096, 0, 0, 4096, 1024, 1024, 1, nullptr, nullptr, nullptr, 0, 0, nullptr, 0, 0, 1, 1);

  // 4) k_sum and kv = kh^T @ vh per (b,h); z = b*8+h, ZD=8
  ksum_kernel<<<256, 256, 0, stream>>>(khT, ksum);
  wmma_gemm_bf16<<<dim3(1, 1, 16), 256, 0, stream>>>(
      khT, 4096, 2048LL, 128LL * 4096, vhT, 4096, 2048LL, 128LL * 4096,
      kvT, 128, 131072LL, 16384LL, 128, 128, 2048, 8,
      nullptr, nullptr, nullptr, 0, 0, nullptr, 0, 0, 1, 1);

  // 5) denom then out = (qh @ kv) / denom -> attn_in (bf16, [tok][H*DH])
  denom_kernel<<<4096, 256, 0, stream>>>(qh, ksum, rden);
  wmma_gemm_bf16<<<dim3(1, 16, 16), 256, 0, stream>>>(
      qh, 1024, 2048LL * 1024, 128LL, kvT, 128, 131072LL, 16384LL,
      attn, 1024, 2048LL * 1024, 128LL, 2048, 128, 128, 8,
      nullptr, nullptr, rden, 16384LL, 2048LL, nullptr, 0, 0, 1, 0);

  // 6) q2 = Q + attn @ Wo + bo  (fp32)
  wmma_gemm_bf16<<<dim3(8, 32, 1), 256, 0, stream>>>(attn, 1024, 0, 0, WoT, 1024, 0, 0,
      q2, 1024, 0, 0, 4096, 1024, 1024, 1, bo, Q, nullptr, 0, 0, nullptr, 0, 0, 0, 0);

  // 7) MoE: LN -> router -> dense experts with gated accumulate
  ln_bf16<<<4096, 256, 0, stream>>>(q2, gm, bm, xm);
  router_kernel<<<512, 256, 0, stream>>>(xm, Wr, br, noise, gts);
  zero_f32<<<16384, 256, 0, stream>>>(mo, 4096 * 1024);
  for (int e = 0; e < 8; ++e) {
    wmma_gemm_bf16<<<dim3(16, 32, 1), 256, 0, stream>>>(
        xm, 1024, 0, 0, W1T + (size_t)e * 2048 * 1024, 1024, 0, 0,
        hbuf, 2048, 0, 0, 4096, 2048, 1024, 1,
        b1 + e * 2048, nullptr, nullptr, 0, 0, nullptr, 0, 2, 1, 0);
    wmma_gemm_bf16<<<dim3(8, 32, 1), 256, 0, stream>>>(
        hbuf, 2048, 0, 0, W2T + (size_t)e * 1024 * 2048, 2048, 0, 0,
        mo, 1024, 0, 0, 4096, 1024, 2048, 1,
        b2 + e * 1024, nullptr, nullptr, 0, 0, gts + e, 8, 0, 0, 0);
  }

  // 8) out = q2 + LN(mo + xm)*gn + bn
  final_kernel<<<4096, 256, 0, stream>>>(mo, xm, q2, gn, bn, out);
}
